// DRNN_75204877353433
// MI455X (gfx1250) — compile-verified, matrix operand order checked by reading, not run on hardware
//
#include <hip/hip_runtime.h>
#include <hip/hip_bf16.h>

// ---------------------------------------------------------------------------
// Problem constants (from reference): B=32, S=512, W=15, E=300, H=256, V=50000,
// C=2, G=3H=768.  BS = B*S = 16384 windows per direction.
// ---------------------------------------------------------------------------
#define B_   32
#define S_   512
#define W_   15
#define E_   300
#define EP_  320          // E padded to multiple of 32 for WMMA K
#define H_   256
#define G_   768
#define C_   2
#define BS_  (B_ * S_)    // 16384
#define HID_ (2 * H_)     // 512

typedef __attribute__((ext_vector_type(16))) _Float16 v16h;
typedef __attribute__((ext_vector_type(8)))  _Float16 v8h;
typedef __attribute__((ext_vector_type(8)))  float    v8f;

union AF { v16h v; v8h h[2]; };

// ---------------------------------------------------------------------------
// WMMA GEMM:  C[M x N] = A[M x K] * B^T  (B row-major [N x K] = torch weight)
//             + bias[n].  K is a template constant (256 / 320 / 512, all
// multiples of 64) so the K loop fully unrolls and the A ping-pong buffers are
// statically indexed -> guaranteed to live in VGPRs (the previous version's
// runtime `buf` index made the compiler spill the buffers to scratch).
//
// Block = 256 threads = 8 waves; each wave owns a 32x64 C tile (2 row-frags x
// 4 col-accumulators = 8 WMMAs per 32-wide K chunk).  Block tile = 256 x 64.
// grid = (N/64, M/256).
//
// B tile (64 x K, shared by all 8 waves) is cooperatively staged in LDS once
// (row stride K+8 when it fits, to spread banks); inner loop reads it with
// ds_load_b128.  A fragments are double-buffered in registers with a 2x
// unrolled software pipeline: chunk kc+32 is fetched before the WMMAs on kc,
// so global latency hides behind 8 WMMAs.
//
// Dual-B: rows [0, Msplit) use B0/bias0, rows >= Msplit use B1/bias1 (lets one
// launch run forward+backward GRU hidden GEMMs together; 16384 % 256 == 0 so
// the switch is block-uniform and EXEC stays all-ones around WMMA).
//
// Fragment layout per CDNA5 ISA 7.12.2 (16-bit A 16x32): lane L holds row
// m = L&15; lane-half h = L>>4 selects K+8h; v16h elems 0..7 = K[8h..8h+7],
// elems 8..15 = K[16+8h..16+8h+7] -> two contiguous v8h (16B) loads.
// C/D (v8f): lane L -> column n = L&15, VGPR r -> row m0 + r + 8*(L>>4).
// ---------------------------------------------------------------------------
template<int K, bool OUT_F16>
__global__ __launch_bounds__(256)
void k_gemm_wmma(const _Float16* __restrict__ A,
                 const _Float16* __restrict__ B0, const _Float16* __restrict__ B1,
                 int Msplit,
                 const float* __restrict__ bias0, const float* __restrict__ bias1,
                 void* __restrict__ Cout, int ldc)
{
    constexpr int LDL = K + ((K < 512) ? 8 : 0);       // padded LDS row stride
    __shared__ __align__(16) _Float16 lds[64 * LDL];   // <= 64 KB (K = 512)

    const int tid  = threadIdx.x;
    const int lane = tid & 31;
    const int wave = tid >> 5;
    const int m0   = blockIdx.y * 256 + wave * 32;
    const int n0   = blockIdx.x * 64;
    const bool lo  = (m0 < Msplit);
    const _Float16* __restrict__ Bm   = lo ? B0 : B1;
    const float*    __restrict__ bias = lo ? bias0 : bias1;
    const int lm   = lane & 15;
    const int half = lane >> 4;

    // ---- cooperative stage of B tile [64 x K] into LDS (v8h granules) ----
    constexpr int CHUNKS = K / 8;
    for (int i = tid; i < 64 * CHUNKS; i += 256) {
        int row = i / CHUNKS, ch = i - row * CHUNKS;
        *(v8h*)&lds[row * LDL + (ch << 3)] =
            *(const v8h*)&Bm[(size_t)(n0 + row) * K + (ch << 3)];
    }
    __syncthreads();

    v8f acc[2][4] = {};
    const _Float16* ap0 = A + (size_t)(m0 + lm) * K + 8 * half;
    const _Float16* ap1 = ap0 + (size_t)16 * K;

    AF a0[2], a1[2];
    a0[0].h[0] = *(const v8h*)(ap0);
    a0[0].h[1] = *(const v8h*)(ap0 + 16);
    a0[1].h[0] = *(const v8h*)(ap1);
    a0[1].h[1] = *(const v8h*)(ap1 + 16);

    // 8 WMMAs (2 row-frags x 4 col-tiles) against LDS-resident B at offset kc
    auto mma8 = [&](const AF (&a)[2], int kc) {
#pragma unroll
        for (int t = 0; t < 4; ++t) {
            AF bf;
            const _Float16* bp = &lds[(t * 16 + lm) * LDL + kc + 8 * half];
            bf.h[0] = *(const v8h*)(bp);
            bf.h[1] = *(const v8h*)(bp + 16);
            acc[0][t] = __builtin_amdgcn_wmma_f32_16x16x32_f16(
                false, a[0].v, false, bf.v, (short)0, acc[0][t], false, false);
            acc[1][t] = __builtin_amdgcn_wmma_f32_16x16x32_f16(
                false, a[1].v, false, bf.v, (short)0, acc[1][t], false, false);
        }
    };

#pragma unroll
    for (int kc = 0; kc < K; kc += 64) {
        // prefetch chunk kc+32 into a1 before consuming a0
        if (kc + 32 < K) {
            a1[0].h[0] = *(const v8h*)(ap0 + kc + 32);
            a1[0].h[1] = *(const v8h*)(ap0 + kc + 48);
            a1[1].h[0] = *(const v8h*)(ap1 + kc + 32);
            a1[1].h[1] = *(const v8h*)(ap1 + kc + 48);
        }
        mma8(a0, kc);
        // prefetch chunk kc+64 into a0 before consuming a1
        if (kc + 64 < K) {
            a0[0].h[0] = *(const v8h*)(ap0 + kc + 64);
            a0[0].h[1] = *(const v8h*)(ap0 + kc + 80);
            a0[1].h[0] = *(const v8h*)(ap1 + kc + 64);
            a0[1].h[1] = *(const v8h*)(ap1 + kc + 80);
        }
        if (kc + 32 < K) mma8(a1, kc + 32);
    }

#pragma unroll
    for (int rt = 0; rt < 2; ++rt) {
#pragma unroll
        for (int t = 0; t < 4; ++t) {
            const int nc = n0 + t * 16 + lm;
            const float bv = bias ? bias[nc] : 0.0f;
#pragma unroll
            for (int r = 0; r < 8; ++r) {
                const int mr = m0 + rt * 16 + r + 8 * half;
                const float val = acc[rt][t][r] + bv;
                if (OUT_F16) ((_Float16*)Cout)[(size_t)mr * ldc + nc] = (_Float16)val;
                else         ((float*)   Cout)[(size_t)mr * ldc + nc] = val;
            }
        }
    }
}

// ---------------------------------------------------------------------------
// Helpers
// ---------------------------------------------------------------------------
__global__ void k_zero(unsigned int* __restrict__ p, long long ndw) {
    long long i = (long long)blockIdx.x * blockDim.x + threadIdx.x;
    if (i < ndw) p[i] = 0u;
}

// float weight [N x K] -> f16 [N x Kpad], zero padded in K
__global__ void k_convert_w(const float* __restrict__ Wf, _Float16* __restrict__ Wh,
                            int N, int K, int Kpad) {
    int idx = blockIdx.x * blockDim.x + threadIdx.x;
    if (idx >= N * Kpad) return;
    int n = idx / Kpad, k = idx - n * Kpad;
    Wh[idx] = (k < K) ? (_Float16)Wf[(size_t)n * K + k] : (_Float16)0.0f;
}

// ex[bs][0..EP) = f16(emb[x[bs]][e]), zero padded e >= E
__global__ void k_gather_emb(const int* __restrict__ x, const float* __restrict__ emb,
                             _Float16* __restrict__ ex) {
    int idx = blockIdx.x * blockDim.x + threadIdx.x;
    if (idx >= BS_ * EP_) return;
    int bs = idx / EP_, e = idx - bs * EP_;
    int tok = x[bs];
    ex[idx] = (e < E_) ? (_Float16)emb[(size_t)tok * E_ + e] : (_Float16)0.0f;
}

// pad token (id 0) projection: out[g] = emb[0]·W_ih[g] + b_ih[g]   (768 vals)
__global__ void k_pad_proj(const float* __restrict__ emb, const float* __restrict__ Wih,
                           const float* __restrict__ bih, float* __restrict__ out) {
    int g = blockIdx.x * blockDim.x + threadIdx.x;
    if (g >= G_) return;
    float s = bih[g];
    for (int e = 0; e < E_; ++e) s += emb[e] * Wih[(size_t)g * E_ + e];
    out[g] = s;
}

// ---------------------------------------------------------------------------
// GRU gate/update, both directions fused.  idx layout [dir][bs][j], j<H.
// forward step t_f uses token q = s + t_f - (W-1); backward step t_b uses
// q = s + t_b; out-of-range q -> constant pad projection.
// ---------------------------------------------------------------------------
__global__ void k_gru_gate(const _Float16* __restrict__ proj_f,
                           const _Float16* __restrict__ proj_b,
                           const float* __restrict__ pad_proj,   // [2][G]
                           const float* __restrict__ hg,         // [2*BS][G]
                           float* __restrict__ h32,              // [2*BS][H]
                           _Float16* __restrict__ h16,
                           int t_f, int t_b) {
    int idx = blockIdx.x * blockDim.x + threadIdx.x;
    if (idx >= 2 * BS_ * H_) return;
    int j   = idx & (H_ - 1);
    int bs  = (idx >> 8) & (BS_ - 1);
    int dir = idx >> 22;                 // 8 (H) + 14 (BS)
    int s   = bs & (S_ - 1);

    int q = (dir == 0) ? (s + t_f - (W_ - 1)) : (s + t_b);
    float xr, xz, xn;
    if (q >= 0 && q < S_) {
        const _Float16* proj = (dir == 0) ? proj_f : proj_b;
        size_t base = (size_t)(bs - s + q) * G_;      // (b*S + q) * G
        xr = (float)proj[base + j];
        xz = (float)proj[base + H_ + j];
        xn = (float)proj[base + 2 * H_ + j];
    } else {
        const float* pp = pad_proj + dir * G_;
        xr = pp[j]; xz = pp[H_ + j]; xn = pp[2 * H_ + j];
    }
    size_t hb = (size_t)(dir * BS_ + bs) * G_;
    float hr = hg[hb + j], hz = hg[hb + H_ + j], hn = hg[hb + 2 * H_ + j];
    float r = 1.0f / (1.0f + __expf(-(xr + hr)));
    float z = 1.0f / (1.0f + __expf(-(xz + hz)));
    float n = tanhf(xn + r * hn);
    float hp = h32[idx];
    float hv = (1.0f - z) * n + z * hp;
    h32[idx] = hv;
    h16[idx] = (_Float16)hv;
}

// hidden[bs][c] = (c<H ? hf : hb)[bs][c&255] * mask(bs)
__global__ void k_hidden(const float* __restrict__ h32, const int* __restrict__ x,
                         float* __restrict__ hidden) {
    int idx = blockIdx.x * blockDim.x + threadIdx.x;
    if (idx >= BS_ * HID_) return;
    int c = idx & (HID_ - 1), bs = idx >> 9;
    float m = (x[bs] > 0) ? 1.0f : 0.0f;
    int dir = c >> 8, j = c & (H_ - 1);
    hidden[idx] = h32[(size_t)(dir * BS_ + bs) * H_ + j] * m;
}

// per-channel mean/var over all 16384 rows (block per channel)
__global__ void k_bnstats(const float* __restrict__ hidden, float* __restrict__ stats) {
    const int c = blockIdx.x;
    float s = 0.f, s2 = 0.f;
    for (int r = threadIdx.x; r < BS_; r += blockDim.x) {
        float v = hidden[(size_t)r * HID_ + c];
        s += v; s2 += v * v;
    }
    __shared__ float sh[256], sh2[256];
    sh[threadIdx.x] = s; sh2[threadIdx.x] = s2;
    __syncthreads();
    for (int off = 128; off > 0; off >>= 1) {
        if (threadIdx.x < off) {
            sh[threadIdx.x]  += sh[threadIdx.x + off];
            sh2[threadIdx.x] += sh2[threadIdx.x + off];
        }
        __syncthreads();
    }
    if (threadIdx.x == 0) {
        float mu = sh[0] / (float)BS_;
        stats[c]        = mu;
        stats[HID_ + c] = sh2[0] / (float)BS_ - mu * mu;
    }
}

// normed16 = f16(((hidden - mu) * rsqrt(var+eps)) * gamma + beta) * mask
__global__ void k_bnnorm(const float* __restrict__ hidden, const float* __restrict__ stats,
                         const float* __restrict__ gamma, const float* __restrict__ beta,
                         const int* __restrict__ x, _Float16* __restrict__ normed16) {
    int idx = blockIdx.x * blockDim.x + threadIdx.x;
    if (idx >= BS_ * HID_) return;
    int c = idx & (HID_ - 1), bs = idx >> 9;
    float mu = stats[c], var = stats[HID_ + c];
    float v = (hidden[idx] - mu) * rsqrtf(var + 1e-5f) * gamma[c] + beta[c];
    float m = (x[bs] > 0) ? 1.0f : 0.0f;
    normed16[idx] = (_Float16)(v * m);
}

// pooled[b][c] = max_s( mlp[bs][c]*mask + (mask-1)*65500 )
__global__ void k_pool(const float* __restrict__ mlp, const int* __restrict__ x,
                       float* __restrict__ pooled) {
    int b = blockIdx.x, c = threadIdx.x;    // 512 threads
    float best = -3.4e38f;
    for (int s = 0; s < S_; ++s) {
        int bs = b * S_ + s;
        float m = (x[bs] > 0) ? 1.0f : 0.0f;
        float v = mlp[(size_t)bs * HID_ + c] * m + (m - 1.0f) * 65500.0f;
        best = fmaxf(best, v);
    }
    pooled[b * HID_ + c] = best;
}

// out[b][k] = pooled[b]·lin_W[k] + lin_b[k]   (64 outputs)
__global__ void k_final(const float* __restrict__ pooled, const float* __restrict__ linW,
                        const float* __restrict__ linb, float* __restrict__ out) {
    int idx = threadIdx.x;
    if (idx >= B_ * C_) return;
    int b = idx >> 1, k = idx & 1;
    float s = linb[k];
    for (int c = 0; c < HID_; ++c) s += pooled[b * HID_ + c] * linW[(size_t)k * HID_ + c];
    out[idx] = s;
}

// ---------------------------------------------------------------------------
// Workspace layout (byte offsets, 256-aligned).  HG (96MB) is dead after the
// recurrence, so hidden / mlp_out / normed16 alias into it.
// ---------------------------------------------------------------------------
static constexpr size_t OFF_EX     = 0;                                   // 16384*320*2
static constexpr size_t OFF_WIHF   = OFF_EX     + (size_t)BS_ * EP_ * 2;  // 768*320*2
static constexpr size_t OFF_WIHB   = OFF_WIHF   + (size_t)G_ * EP_ * 2;
static constexpr size_t OFF_WHHF   = OFF_WIHB   + (size_t)G_ * EP_ * 2;   // 768*256*2
static constexpr size_t OFF_WHHB   = OFF_WHHF   + (size_t)G_ * H_ * 2;
static constexpr size_t OFF_MLPW   = OFF_WHHB   + (size_t)G_ * H_ * 2;    // 512*512*2
static constexpr size_t OFF_PROJF  = OFF_MLPW   + (size_t)HID_ * HID_ * 2;
static constexpr size_t OFF_PROJB  = OFF_PROJF  + (size_t)BS_ * G_ * 2;
static constexpr size_t OFF_PADPRJ = OFF_PROJB  + (size_t)BS_ * G_ * 2;   // 2*768*4
static constexpr size_t OFF_H32    = OFF_PADPRJ + 8192;
static constexpr size_t OFF_H16    = OFF_H32    + (size_t)2 * BS_ * H_ * 4;
static constexpr size_t OFF_HG     = OFF_H16    + (size_t)2 * BS_ * H_ * 2;
static constexpr size_t OFF_HIDDEN = OFF_HG;                              // alias
static constexpr size_t OFF_MLPOUT = OFF_HG + (size_t)BS_ * HID_ * 4;     // alias
static constexpr size_t OFF_NORM16 = OFF_MLPOUT + (size_t)BS_ * HID_ * 4; // alias
static constexpr size_t OFF_STATS  = OFF_HG + (size_t)2 * BS_ * G_ * 4;
static constexpr size_t OFF_POOLED = OFF_STATS + 4096;

extern "C" void kernel_launch(void* const* d_in, const int* in_sizes, int n_in,
                              void* d_out, int out_size, void* d_ws, size_t ws_size,
                              hipStream_t stream) {
    (void)in_sizes; (void)n_in; (void)out_size; (void)ws_size;
    const int*   x      = (const int*)  d_in[0];
    const float* emb    = (const float*)d_in[5];
    const float* W_ih_f = (const float*)d_in[6];
    const float* W_hh_f = (const float*)d_in[7];
    const float* b_ih_f = (const float*)d_in[8];
    const float* b_hh_f = (const float*)d_in[9];
    const float* W_ih_b = (const float*)d_in[10];
    const float* W_hh_b = (const float*)d_in[11];
    const float* b_ih_b = (const float*)d_in[12];
    const float* b_hh_b = (const float*)d_in[13];
    const float* gamma  = (const float*)d_in[14];
    const float* beta   = (const float*)d_in[15];
    const float* mlp_W  = (const float*)d_in[16];
    const float* mlp_b  = (const float*)d_in[17];
    const float* lin_W  = (const float*)d_in[18];
    const float* lin_b  = (const float*)d_in[19];
    float* out = (float*)d_out;

    char* ws = (char*)d_ws;
    _Float16* ex     = (_Float16*)(ws + OFF_EX);
    _Float16* wihf   = (_Float16*)(ws + OFF_WIHF);
    _Float16* wihb   = (_Float16*)(ws + OFF_WIHB);
    _Float16* whhf   = (_Float16*)(ws + OFF_WHHF);
    _Float16* whhb   = (_Float16*)(ws + OFF_WHHB);
    _Float16* mlpw   = (_Float16*)(ws + OFF_MLPW);
    _Float16* projf  = (_Float16*)(ws + OFF_PROJF);
    _Float16* projb  = (_Float16*)(ws + OFF_PROJB);
    float*    padprj = (float*)   (ws + OFF_PADPRJ);
    float*    h32    = (float*)   (ws + OFF_H32);
    _Float16* h16    = (_Float16*)(ws + OFF_H16);
    float*    hg     = (float*)   (ws + OFF_HG);
    float*    hidden = (float*)   (ws + OFF_HIDDEN);
    float*    mlpout = (float*)   (ws + OFF_MLPOUT);
    _Float16* norm16 = (_Float16*)(ws + OFF_NORM16);
    float*    stats  = (float*)   (ws + OFF_STATS);
    float*    pooled = (float*)   (ws + OFF_POOLED);

    const int T = 256;
    auto blocks = [](long long n, int t) { return (unsigned)((n + t - 1) / t); };

    // 0) zero h32 + h16 (contiguous region)
    {
        long long ndw = ((long long)2 * BS_ * H_ * 4 + (long long)2 * BS_ * H_ * 2) / 4;
        k_zero<<<blocks(ndw, T), T, 0, stream>>>((unsigned int*)(ws + OFF_H32), ndw);
    }
    // 1) weight conversions (f32 -> f16 padded)
    k_convert_w<<<blocks((long long)G_ * EP_, T), T, 0, stream>>>(W_ih_f, wihf, G_, E_, EP_);
    k_convert_w<<<blocks((long long)G_ * EP_, T), T, 0, stream>>>(W_ih_b, wihb, G_, E_, EP_);
    k_convert_w<<<blocks((long long)G_ * H_,  T), T, 0, stream>>>(W_hh_f, whhf, G_, H_, H_);
    k_convert_w<<<blocks((long long)G_ * H_,  T), T, 0, stream>>>(W_hh_b, whhb, G_, H_, H_);
    k_convert_w<<<blocks((long long)HID_ * HID_, T), T, 0, stream>>>(mlp_W, mlpw, HID_, HID_, HID_);
    // 2) embedding gather + pad-token projections
    k_gather_emb<<<blocks((long long)BS_ * EP_, T), T, 0, stream>>>(x, emb, ex);
    k_pad_proj<<<blocks(G_, T), T, 0, stream>>>(emb, W_ih_f, b_ih_f, padprj);
    k_pad_proj<<<blocks(G_, T), T, 0, stream>>>(emb, W_ih_b, b_ih_b, padprj + G_);
    // 3) input projection GEMMs (shared token embeddings, per-direction weights)
    {
        dim3 grid(G_ / 64, BS_ / 256);
        k_gemm_wmma<EP_, true><<<grid, 256, 0, stream>>>(ex, wihf, wihf, 1 << 30,
                                                         b_ih_f, b_ih_f, projf, G_);
        k_gemm_wmma<EP_, true><<<grid, 256, 0, stream>>>(ex, wihb, wihb, 1 << 30,
                                                         b_ih_b, b_ih_b, projb, G_);
    }
    // 4) 15 recurrent steps; all 2*16384 windows advance in lockstep
    for (int i = 0; i < W_; ++i) {
        dim3 grid(G_ / 64, (2 * BS_) / 256);
        k_gemm_wmma<H_, false><<<grid, 256, 0, stream>>>(h16, whhf, whhb, BS_,
                                                         b_hh_f, b_hh_b, hg, G_);
        k_gru_gate<<<blocks((long long)2 * BS_ * H_, T), T, 0, stream>>>(
            projf, projb, padprj, hg, h32, h16, /*t_f=*/i, /*t_b=*/(W_ - 1) - i);
    }
    // 5) concat+mask, batchnorm (batch stats), normalize -> f16
    k_hidden<<<blocks((long long)BS_ * HID_, T), T, 0, stream>>>(h32, x, hidden);
    k_bnstats<<<HID_, 256, 0, stream>>>(hidden, stats);
    k_bnnorm<<<blocks((long long)BS_ * HID_, T), T, 0, stream>>>(hidden, stats, gamma, beta, x, norm16);
    // 6) MLP GEMM
    {
        dim3 grid(HID_ / 64, BS_ / 256);
        k_gemm_wmma<HID_, false><<<grid, 256, 0, stream>>>(norm16, mlpw, mlpw, 1 << 30,
                                                           mlp_b, mlp_b, mlpout, HID_);
    }
    // 7) masked max-pool + final linear
    k_pool<<<B_, HID_, 0, stream>>>(mlpout, x, pooled);
    k_final<<<1, 64, 0, stream>>>(pooled, lin_W, lin_b, out);
}
